// CausalSelfAttention_66331474920217
// MI455X (gfx1250) — matrix operand order repaired and checked by reading, not compiled, not run on hardware
//
#include <hip/hip_runtime.h>

// ---------------------------------------------------------------------------
// CausalSelfAttention forward for MI455X (gfx1250), wave32 + WMMA bf16,
// TDM (tensor_load_to_lds) staging for GEMM weights, async global->LDS
// staging for attention K/V tiles.
// B=4, T=2048, C=1024, H=16, hs=64.
// ---------------------------------------------------------------------------

typedef __attribute__((ext_vector_type(16))) __bf16 v16bf;
typedef __attribute__((ext_vector_type(8)))  __bf16 v8bf;
typedef __attribute__((ext_vector_type(8)))  float  v8f;
typedef __attribute__((ext_vector_type(4)))  unsigned v4u;
typedef __attribute__((ext_vector_type(8)))  int   v8i;
typedef __attribute__((ext_vector_type(4)))  int   v4i;

#define BATCH 4
#define SEQ   2048
#define CDIM  1024
#define NHEAD 16
#define HS    64
#define MROWS (BATCH*SEQ)   /* 8192 */

static __device__ __forceinline__ v16bf cat16(v8bf lo, v8bf hi) {
  return __builtin_shufflevector(lo, hi, 0,1,2,3,4,5,6,7,8,9,10,11,12,13,14,15);
}

// A-fragment (16x32 bf16, MxK). lanes 0-15: row M=l, K=0..15;
// lanes 16-31: row M=l, K=16..31.
static __device__ __forceinline__ v16bf load_afrag(const __bf16* row, int k0, int half) {
  v8bf lo = *(const v8bf*)(row + k0 + half*16);
  v8bf hi = *(const v8bf*)(row + k0 + half*16 + 8);
  return cat16(lo, hi);
}

// B-fragment (32x16 bf16, KxN) from a pointer to the 32 K-contiguous elements
// of column n (lane l = column l; half 0: K=0..15, half 1: K=16..31).
static __device__ __forceinline__ v16bf load_bfrag(const __bf16* col, int k0, int half) {
  v8bf lo = *(const v8bf*)(col + k0 + half*16);
  v8bf hi = *(const v8bf*)(col + k0 + half*16 + 8);
  return cat16(lo, hi);
}

static __device__ __forceinline__ v8f wmma_bf16(v16bf a, v16bf b, v8f c) {
  return __builtin_amdgcn_wmma_f32_16x16x32_bf16(false, a, false, b, (short)0, c,
                                                 false, false);
}

// Async 16B global -> LDS copy (ASYNCcnt-tracked).
static __device__ __forceinline__ void async_cp16(unsigned lds_addr, const void* gptr) {
  unsigned long long ga = (unsigned long long)(uintptr_t)gptr;
  asm volatile("global_load_async_to_lds_b128 %0, %1, off"
               :: "v"(lds_addr), "v"(ga) : "memory");
}

// TDM: stage a [64 rows x 32 K] bf16 tile (row stride 1024 elems) into LDS.
static __device__ __forceinline__ void tdm_load_tile(unsigned lds_addr, const __bf16* gptr) {
  unsigned long long ga = (unsigned long long)(uintptr_t)gptr;
  v4u g0;
  g0.x = 1u;                                      // count=1, no gather
  g0.y = lds_addr;                                // lds_addr [63:32]
  g0.z = (unsigned)(ga & 0xffffffffu);            // global_addr [95:64]
  g0.w = (unsigned)((ga >> 32) & 0x01ffffffu) | (2u << 30);   // addr hi | type=2
  v8i g1;
  g1[0] = 0x00010000;            // data_size=1 (2 bytes); mask/flags 0
  g1[1] = (int)(1024u << 16);    // tensor_dim0[15:0] = 1024  (bits 79:48)
  g1[2] = (int)(1024u << 16);    // tensor_dim0 hi=0 | tensor_dim1[15:0]=1024
  g1[3] = (int)(32u << 16);      // tensor_dim1 hi=0 | tile_dim0 = 32
  g1[4] = 64;                    // tile_dim1 = 64 | tile_dim2 = 0
  g1[5] = 1024;                  // tensor_dim0_stride[31:0] = 1024 elems
  g1[6] = 0;                     // stride hi | tensor_dim1_stride lo
  g1[7] = 0;
  v4i z4 = {0, 0, 0, 0};
#if defined(__clang_major__) && (__clang_major__ >= 23)
  v8i z8 = {0, 0, 0, 0, 0, 0, 0, 0};
  __builtin_amdgcn_tensor_load_to_lds(g0, g1, z4, z4, z8, 0);
#else
  __builtin_amdgcn_tensor_load_to_lds(g0, g1, z4, z4, 0);
#endif
}

// ---------------------------------------------------------------------------
// Prep kernels
// ---------------------------------------------------------------------------
__global__ void cvt_f32_bf16(const float* __restrict__ in, __bf16* __restrict__ out, int n) {
  int i = blockIdx.x * blockDim.x + threadIdx.x;
  if (i < n) out[i] = (__bf16)in[i];
}

// Wt[n*C + k] = (bf16) W[k*C + n]
__global__ void transpose_w(const float* __restrict__ W, __bf16* __restrict__ Wt) {
  int i = blockIdx.x * blockDim.x + threadIdx.x;
  int n = i >> 10;
  int k = i & (CDIM - 1);
  Wt[i] = (__bf16)W[(size_t)k * CDIM + n];
}

// vT[((b*H + h)*64 + d)*T + t] = v[(b*T + t)*C + h*64 + d]
__global__ void transpose_v_kernel(const __bf16* __restrict__ v, __bf16* __restrict__ vT) {
  int i = blockIdx.x * blockDim.x + threadIdx.x;
  int d  = i & (HS - 1);
  int h  = (i >> 6) & (NHEAD - 1);
  int bt = i >> 10;
  int t  = bt & (SEQ - 1);
  int b  = bt >> 11;
  vT[(((size_t)((b*NHEAD + h)*HS + d)) << 11) + t] = v[i];
}

// ---------------------------------------------------------------------------
// WMMA GEMM: out[m][n] = sum_k A[m][k]*Bt[n][k] + bias[n]
// Block = 8 waves -> 128x64 output tile. Per K-step the 64x32 Bt tile is
// staged into LDS by the Tensor Data Mover (double-buffered); each wave
// reuses one A-fragment across 4 WMMAs.
// ---------------------------------------------------------------------------
__global__ void __launch_bounds__(256)
gemm_wmma(const __bf16* __restrict__ A, const __bf16* __restrict__ Bt,
          const float* __restrict__ bias,
          __bf16* __restrict__ outBf, float* __restrict__ outF,
          int M, int N, int K) {
  __shared__ __align__(16) __bf16 btile[2][64 * 32];   // [n-row][K] per buffer
  const int wave = threadIdx.x >> 5;
  const int ln   = threadIdx.x & 31;
  const int half = ln >> 4;
  const int l    = ln & 15;
  const int mt   = blockIdx.x * 8 + wave;   // 16-row tile
  const int nb   = blockIdx.y;              // 64-column block

  const __bf16* Arow  = A  + (size_t)(mt*16 + l) * K;
  const __bf16* Bbase = Bt + (size_t)(nb*64) * K;      // 64 rows of Bt

  if (threadIdx.x == 0)
    tdm_load_tile((unsigned)(uintptr_t)&btile[0][0], Bbase + 0);

  v8f acc[4] = {{}, {}, {}, {}};
  for (int k0 = 0; k0 < K; k0 += 32) {
    const int cur = (k0 >> 5) & 1;
    if (threadIdx.x == 0) {
      if (k0 + 32 < K) {
        tdm_load_tile((unsigned)(uintptr_t)&btile[cur ^ 1][0], Bbase + (k0 + 32));
        __builtin_amdgcn_s_wait_tensorcnt((short)1);
      } else {
        __builtin_amdgcn_s_wait_tensorcnt((short)0);
      }
    }
    __syncthreads();                       // staged tile visible to all waves
    v16bf a = load_afrag(Arow, k0, half);
#pragma unroll
    for (int s = 0; s < 4; ++s) {
      const __bf16* col = &btile[cur][(s*16 + l) * 32];
      v16bf b = cat16(*(const v8bf*)(col + half*16),
                      *(const v8bf*)(col + half*16 + 8));
      acc[s] = wmma_bf16(a, b, acc[s]);
    }
    __syncthreads();                       // reads done before next TDM overwrite
  }

#pragma unroll
  for (int s = 0; s < 4; ++s) {
    const int n  = nb*64 + s*16 + l;
    const float bb = bias[n];
#pragma unroll
    for (int r = 0; r < 8; ++r) {
      const int m = mt*16 + r + half*8;     // D layout: VGPR r -> row r+8*half
      const float valr = acc[s][r] + bb;
      if (outF) outF[(size_t)m * N + n] = valr;
      else      outBf[(size_t)m * N + n] = (__bf16)valr;
    }
  }
}

// ---------------------------------------------------------------------------
// Flash attention: block = 8 waves = 128 consecutive queries of one (b,h).
// Key blocks of 32; K and V tiles staged cooperatively with async
// global->LDS b128 copies, double-buffered (ASYNCcnt).
// ---------------------------------------------------------------------------
__global__ void __launch_bounds__(256)
attn_kernel(const __bf16* __restrict__ Q, const __bf16* __restrict__ Kc,
            const __bf16* __restrict__ Vt, __bf16* __restrict__ Y) {
  __shared__ __align__(16) __bf16 ktile[2][32 * 64];   // [key][d]
  __shared__ __align__(16) __bf16 vtile[2][64 * 32];   // [d][key]
  __shared__ __align__(16) __bf16 plds[8 * 16 * 32];   // P bounce, 1KB/wave
  const int tid  = threadIdx.x;
  const int wave = tid >> 5;
  const int ln   = tid & 31;
  const int half = ln >> 4;
  const int l    = ln & 15;

  const int blk = blockIdx.x;              // 0..1023
  const int qtb = blk & 15;                // 128-query block within sequence
  const int h   = (blk >> 4) & (NHEAD - 1);
  const int b   = blk >> 8;
  const int qbaseB = qtb * 128;
  const int qbase  = qbaseB + wave * 16;
  const size_t hoff = (size_t)h * HS;

  const __bf16* Kbh  = Kc + (size_t)(b*SEQ) * CDIM + hoff;
  const __bf16* Vtbh = Vt + (((size_t)((b*NHEAD + h) * HS)) << 11);

  // cooperative async staging of one key-block (32 keys): each thread 2x16B
  auto issue_copy = [&](int buf, int kbase) {
    {
      const int krow = tid >> 3, kcol = (tid & 7) * 8;      // 8 lanes x 16B = 128B/row
      unsigned lk = (unsigned)(uintptr_t)&ktile[buf][0] + (unsigned)(krow*128 + kcol*2);
      async_cp16(lk, Kbh + (size_t)(kbase + krow) * CDIM + kcol);
    }
    {
      const int vrow = tid >> 2, vcol = (tid & 3) * 8;      // 4 lanes x 16B = 64B/row
      unsigned lv = (unsigned)(uintptr_t)&vtile[buf][0] + (unsigned)(vrow*64 + vcol*2);
      async_cp16(lv, Vtbh + ((size_t)vrow << 11) + kbase + vcol);
    }
  };

  const __bf16* qrow = Q + ((size_t)(b*SEQ + qbase + l)) * CDIM + hoff;
  const v16bf qa0 = load_afrag(qrow, 0,  half);
  const v16bf qa1 = load_afrag(qrow, 32, half);

  v8f o0 = {}, o1 = {}, o2 = {}, o3 = {};
  float mi[8], li[8];
#pragma unroll
  for (int r = 0; r < 8; ++r) { mi[r] = -1e30f; li[r] = 0.0f; }

  __bf16* mylds = plds + wave * (16 * 32);
  const float scale = 0.125f;                   // 1/sqrt(64)
  const int kbmaxB = (qbaseB + 127) >> 5;       // block-uniform trip count
  const int kbmaxW = (qbase + 15) >> 5;         // this wave's causal limit

  issue_copy(0, 0);
  for (int kb = 0; kb <= kbmaxB; ++kb) {
    const int cur = kb & 1;
    const int kbase = kb * 32;
    if (kb < kbmaxB) {
      issue_copy(cur ^ 1, kbase + 32);
      asm volatile("s_wait_asynccnt 0x2" ::: "memory");   // current tile done
    } else {
      asm volatile("s_wait_asynccnt 0x0" ::: "memory");
    }
    __syncthreads();                                      // tiles visible block-wide

    if (kb <= kbmaxW) {
      // --- S = Q K^T from LDS K tile (column n = key row, contiguous d) ---
      v8f s0 = {}, s1 = {};
      const __bf16* kc0 = &ktile[cur][l * 64];
      const __bf16* kc1 = &ktile[cur][(16 + l) * 64];
      s0 = wmma_bf16(qa0, load_bfrag(kc0, 0,  half), s0);
      s0 = wmma_bf16(qa1, load_bfrag(kc0, 32, half), s0);
      s1 = wmma_bf16(qa0, load_bfrag(kc1, 0,  half), s1);
      s1 = wmma_bf16(qa1, load_bfrag(kc1, 32, half), s1);

      const bool needMask = (kbase + 32 > qbase);
      float p0[8], p1[8];
#pragma unroll
      for (int r = 0; r < 8; ++r) {
        float a0 = s0[r] * scale;
        float a1 = s1[r] * scale;
        if (needMask) {
          const int row = qbase + r + half*8;
          if (kbase + l      > row) a0 = -1e30f;
          if (kbase + 16 + l > row) a1 = -1e30f;
        }
        p0[r] = a0; p1[r] = a1;
      }

      // --- online softmax (row across 16 lanes of this half-wave) ---
#pragma unroll
      for (int r = 0; r < 8; ++r) {
        float mx = fmaxf(p0[r], p1[r]);
#pragma unroll
        for (int msk = 8; msk >= 1; msk >>= 1) mx = fmaxf(mx, __shfl_xor(mx, msk, 32));
        const float mnew  = fmaxf(mi[r], mx);
        const float alpha = __expf(mi[r] - mnew);
        const float e0 = __expf(p0[r] - mnew);
        const float e1 = __expf(p1[r] - mnew);
        float rs = e0 + e1;
#pragma unroll
        for (int msk = 8; msk >= 1; msk >>= 1) rs += __shfl_xor(rs, msk, 32);
        li[r] = li[r] * alpha + rs;
        mi[r] = mnew;
        o0[r] *= alpha; o1[r] *= alpha; o2[r] *= alpha; o3[r] *= alpha;
        p0[r] = e0; p1[r] = e1;
      }

      // --- P (D layout) -> LDS row-major -> bf16 A fragment ---
#pragma unroll
      for (int r = 0; r < 8; ++r) {
        const int m = r + half*8;
        mylds[m*32 + l]      = (__bf16)p0[r];
        mylds[m*32 + 16 + l] = (__bf16)p1[r];
      }
      asm volatile("s_wait_dscnt 0x0" ::: "memory");      // wave-private RAW fence
      const v16bf pa = cat16(*(const v8bf*)(mylds + l*32 + half*16),
                             *(const v8bf*)(mylds + l*32 + half*16 + 8));

      // --- O += P @ V from LDS V tile (column d, contiguous keys) ---
      o0 = wmma_bf16(pa, load_bfrag(&vtile[cur][( 0 + l)*32], 0, half), o0);
      o1 = wmma_bf16(pa, load_bfrag(&vtile[cur][(16 + l)*32], 0, half), o1);
      o2 = wmma_bf16(pa, load_bfrag(&vtile[cur][(32 + l)*32], 0, half), o2);
      o3 = wmma_bf16(pa, load_bfrag(&vtile[cur][(48 + l)*32], 0, half), o3);
    }
    __syncthreads();                       // reads done before buffer reuse
  }

  // --- normalize and store y (bf16, [B*T][C] at head offset) ---
#pragma unroll
  for (int r = 0; r < 8; ++r) {
    const int m = qbase + r + half*8;
    const float inv = 1.0f / li[r];
    __bf16* yrow = Y + ((size_t)(b*SEQ + m)) * CDIM + hoff;
    yrow[ 0 + l] = (__bf16)(o0[r] * inv);
    yrow[16 + l] = (__bf16)(o1[r] * inv);
    yrow[32 + l] = (__bf16)(o2[r] * inv);
    yrow[48 + l] = (__bf16)(o3[r] * inv);
  }
}

// ---------------------------------------------------------------------------
extern "C" void kernel_launch(void* const* d_in, const int* in_sizes, int n_in,
                              void* d_out, int out_size, void* d_ws, size_t ws_size,
                              hipStream_t stream) {
  const float* x  = (const float*)d_in[0];
  const float* Wq = (const float*)d_in[1];
  const float* bq = (const float*)d_in[2];
  const float* Wk = (const float*)d_in[3];
  const float* bk = (const float*)d_in[4];
  const float* Wv = (const float*)d_in[5];
  const float* bv = (const float*)d_in[6];
  float* out = (float*)d_out;

  const size_t NX = (size_t)MROWS * CDIM;
  const size_t NW = (size_t)CDIM * CDIM;
  __bf16* ws  = (__bf16*)d_ws;
  __bf16* xbf = ws;            ws += NX;
  __bf16* WqT = ws;            ws += NW;
  __bf16* WkT = ws;            ws += NW;
  __bf16* WvT = ws;            ws += NW;
  __bf16* qws = ws;            ws += NX;
  __bf16* kws = ws;            ws += NX;
  __bf16* vws = ws;            ws += NX;
  __bf16* vTs = ws;            ws += NX;
  __bf16* yws = ws;            ws += NX;

  cvt_f32_bf16<<<(int)(NX/256), 256, 0, stream>>>(x, xbf, (int)NX);
  transpose_w<<<(int)(NW/256), 256, 0, stream>>>(Wq, WqT);
  transpose_w<<<(int)(NW/256), 256, 0, stream>>>(Wk, WkT);
  transpose_w<<<(int)(NW/256), 256, 0, stream>>>(Wv, WvT);

  dim3 ggrid(MROWS/128, CDIM/64);   // 64 x 16 blocks, 8 waves each
  gemm_wmma<<<ggrid, 256, 0, stream>>>(xbf, WqT, bq, qws, nullptr, MROWS, CDIM, CDIM);
  gemm_wmma<<<ggrid, 256, 0, stream>>>(xbf, WkT, bk, kws, nullptr, MROWS, CDIM, CDIM);
  gemm_wmma<<<ggrid, 256, 0, stream>>>(xbf, WvT, bv, vws, nullptr, MROWS, CDIM, CDIM);

  transpose_v_kernel<<<(int)(NX/256), 256, 0, stream>>>(vws, vTs);
  attn_kernel<<<(BATCH*NHEAD*(SEQ/128)), 256, 0, stream>>>(qws, kws, vTs, yws);
  gemm_wmma<<<ggrid, 256, 0, stream>>>(yws, WvT, bv, nullptr, out, MROWS, CDIM, CDIM);
}
